// CausalSelfAttention_26594437497017
// MI455X (gfx1250) — matrix operand order repaired and checked by reading, not compiled, not run on hardware
//
#include <hip/hip_runtime.h>
#include <math.h>
#include <stdint.h>

typedef __attribute__((ext_vector_type(16))) _Float16 v16h;
typedef __attribute__((ext_vector_type(4))) _Float16 v4h;
typedef __attribute__((ext_vector_type(8))) float v8f;

#define B_ 4
#define T_ 2048
#define C_ 1024
#define H_ 16
#define D_ 64
#define N3C (3 * C_)

union F16Frag {
    v16h v;
    uint4 q[2];
};

static __device__ inline v8f zero8() {
    v8f z;
#pragma unroll
    for (int i = 0; i < 8; ++i) z[i] = 0.0f;
    return z;
}

static __device__ inline v8f wmma16(v16h a, v16h b, v8f c) {
    // D(16x16 f32) = A(16x32 f16) * B(32x16 f16) + C
    return __builtin_amdgcn_wmma_f32_16x16x32_f16(
        /*neg_a=*/false, a, /*neg_b=*/false, b,
        /*c_mod=*/(short)0, c, /*reuse_a=*/false, /*reuse_b=*/false);
}

// CDNA5 async copy: 16B per lane, global -> LDS, tracked by ASYNCcnt.
static __device__ __forceinline__ void async_b128_to_lds(uint32_t lds_off,
                                                         const void* gsrc) {
    asm volatile("global_load_async_to_lds_b128 %0, %1, off" ::"v"(lds_off),
                 "v"((unsigned long long)(uintptr_t)gsrc)
                 : "memory");
}

static __device__ __forceinline__ void wait_async0() {
    asm volatile("s_wait_asynccnt 0" ::: "memory");
}

// ---------------------------------------------------------------------------
// Conversion kernels (one-time, memory-bound).
// ---------------------------------------------------------------------------
__global__ __launch_bounds__(256) void cvt_f16_kernel(
    const float* __restrict__ src, _Float16* __restrict__ dst) {
    size_t i = (size_t)blockIdx.x * 256 + threadIdx.x;
    float4 v = ((const float4*)src)[i];
    v4h o;
    o[0] = (_Float16)v.x;
    o[1] = (_Float16)v.y;
    o[2] = (_Float16)v.z;
    o[3] = (_Float16)v.w;
    ((v4h*)dst)[i] = o;
}

// W f32 [K][N] -> WT f16 [N][K] (64x64 LDS tile transpose)
__global__ __launch_bounds__(256) void cvt_transpose_kernel(
    const float* __restrict__ W, _Float16* __restrict__ WT, int K, int N) {
    __shared__ float tile[64][65];
    const int nTiles = N >> 6;
    const int kb = (blockIdx.x / nTiles) << 6;
    const int nb = (blockIdx.x % nTiles) << 6;
    const int t = threadIdx.x;

    const int r = t >> 4;          // 0..15
    const int c4 = (t & 15) * 4;   // 0..60
#pragma unroll
    for (int rr = 0; rr < 64; rr += 16) {
        float4 v = *(const float4*)&W[(size_t)(kb + r + rr) * N + nb + c4];
        tile[r + rr][c4 + 0] = v.x;
        tile[r + rr][c4 + 1] = v.y;
        tile[r + rr][c4 + 2] = v.z;
        tile[r + rr][c4 + 3] = v.w;
    }
    __syncthreads();

    const int rn = t >> 2;         // 0..63 (n within tile)
    const int ck = (t & 3) * 16;   // k base
    F16Frag f;
#pragma unroll
    for (int i = 0; i < 16; ++i) f.v[i] = (_Float16)tile[ck + i][rn];
    _Float16* dp = WT + (size_t)(nb + rn) * K + kb + ck;
    *(uint4*)(dp) = f.q[0];
    *(uint4*)(dp + 8) = f.q[1];
}

// ---------------------------------------------------------------------------
// f16 GEMM mainloop: block tile 256(M) x 64(N), K=1024.
// A row-major f16; BT transposed f16 [N][1024]. 8 waves share a double-
// buffered LDS B panel (64 n-rows x 64 k halves = 8KB) streamed with async
// b128 copies. Each wave owns 32 M-rows (two row groups), so every B
// fragment read from LDS feeds two WMMAs (halved LDS traffic per wmma).
// 16 f16-WMMAs per 64-k chunk per wave.
// ---------------------------------------------------------------------------
static __device__ inline void gemm_f16_tile32(const _Float16* __restrict__ A16,
                                              const _Float16* __restrict__ BT16,
                                              int m0_wave, int n0,
                                              _Float16* Bs, v8f acc[2][4]) {
    const int tid = threadIdx.x;
    const int lane = tid & 31;
    const int lane16 = lane & 15;
    const int half = lane >> 4;

    // Per-thread slice of B chunk copy: 64 rows x 64 halves, 32B/thread.
    const int nr = tid >> 2;           // 0..63
    const int kcb = (tid & 3) * 16;    // 0,16,32,48 halves
    const _Float16* gsrc = BT16 + (size_t)(n0 + nr) * 1024 + kcb;
    const uint32_t lbase =
        (uint32_t)(uintptr_t)Bs + (uint32_t)(nr * 64 + kcb) * 2u;

    const _Float16* arow0 = A16 + (size_t)(m0_wave + lane16) * 1024 + half * 16;
    const _Float16* arow1 = arow0 + (size_t)16 * 1024;

    async_b128_to_lds(lbase, gsrc);
    async_b128_to_lds(lbase + 16, gsrc + 8);
    wait_async0();
    __syncthreads();

    int buf = 0;
    for (int kc = 0; kc < 1024; kc += 64) {
        if (kc + 64 < 1024) {
            const _Float16* gn = gsrc + kc + 64;
            uint32_t ln = lbase + (uint32_t)((buf ^ 1) * 8192);
            async_b128_to_lds(ln, gn);
            async_b128_to_lds(ln + 16, gn + 8);
        }
        const _Float16* bs = Bs + buf * 4096;
#pragma unroll
        for (int s = 0; s < 2; ++s) {
            const int kk = kc + 32 * s;
            // Independent fragment loads first (lets the scheduler hoist
            // all ds/global loads ahead of the WMMA burst).
            F16Frag a0, a1;
            a0.q[0] = *(const uint4*)(arow0 + kk);
            a0.q[1] = *(const uint4*)(arow0 + kk + 8);
            a1.q[0] = *(const uint4*)(arow1 + kk);
            a1.q[1] = *(const uint4*)(arow1 + kk + 8);
            F16Frag b0, b1, b2, b3;
            const _Float16* bp = bs + (size_t)lane16 * 64 + 32 * s + half * 16;
            b0.q[0] = *(const uint4*)(bp);
            b0.q[1] = *(const uint4*)(bp + 8);
            b1.q[0] = *(const uint4*)(bp + 16 * 64);
            b1.q[1] = *(const uint4*)(bp + 16 * 64 + 8);
            b2.q[0] = *(const uint4*)(bp + 32 * 64);
            b2.q[1] = *(const uint4*)(bp + 32 * 64 + 8);
            b3.q[0] = *(const uint4*)(bp + 48 * 64);
            b3.q[1] = *(const uint4*)(bp + 48 * 64 + 8);

            acc[0][0] = wmma16(a0.v, b0.v, acc[0][0]);
            acc[1][0] = wmma16(a1.v, b0.v, acc[1][0]);
            acc[0][1] = wmma16(a0.v, b1.v, acc[0][1]);
            acc[1][1] = wmma16(a1.v, b1.v, acc[1][1]);
            acc[0][2] = wmma16(a0.v, b2.v, acc[0][2]);
            acc[1][2] = wmma16(a1.v, b2.v, acc[1][2]);
            acc[0][3] = wmma16(a0.v, b3.v, acc[0][3]);
            acc[1][3] = wmma16(a1.v, b3.v, acc[1][3]);
        }
        wait_async0();
        __syncthreads();
        buf ^= 1;
    }
}

// ---------------------------------------------------------------------------
// Kernel: qkv = x16 @ W_attn  -> q16 (scaled by 1/sqrt(D), [B,H,T,D]),
//                               k16 ([B,H,T,D]), vt16 (transposed [B,H,D,T]).
// ---------------------------------------------------------------------------
__global__ __launch_bounds__(256) void qkv_gemm_f16(
    const _Float16* __restrict__ x16, const _Float16* __restrict__ WaT,
    _Float16* __restrict__ q16, _Float16* __restrict__ k16,
    _Float16* __restrict__ vt16) {
    __shared__ _Float16 Bs[2 * 4096];

    const int lane = threadIdx.x & 31;
    const int wave = threadIdx.x >> 5;
    const int lane16 = lane & 15;
    const int half = lane >> 4;

    const int NB = N3C / 64;  // 48
    const int m0 = (blockIdx.x / NB) * 256 + wave * 32;
    const int n0 = (blockIdx.x % NB) * 64;

    v8f acc[2][4];
#pragma unroll
    for (int g = 0; g < 2; ++g)
#pragma unroll
        for (int j = 0; j < 4; ++j) acc[g][j] = zero8();

    gemm_f16_tile32(x16, WaT, m0, n0, Bs, acc);

#pragma unroll
    for (int g = 0; g < 2; ++g) {
#pragma unroll
        for (int j = 0; j < 4; ++j) {
#pragma unroll
            for (int r = 0; r < 8; ++r) {
                int row = m0 + g * 16 + r + 8 * half;  // token row
                int col = n0 + 16 * j + lane16;        // [0, 3C)
                int b = row >> 11;
                int t = row & (T_ - 1);
                int which = col >> 10;
                int c = col & (C_ - 1);
                int h = c >> 6;
                int d = c & (D_ - 1);
                float val = acc[g][j][r];
                size_t bhbase = (size_t)(b * H_ + h);
                if (which == 0)
                    q16[(bhbase * T_ + t) * D_ + d] = (_Float16)(val * 0.125f);
                else if (which == 1)
                    k16[(bhbase * T_ + t) * D_ + d] = (_Float16)val;
                else
                    vt16[(bhbase * D_ + d) * T_ + t] = (_Float16)val;
            }
        }
    }
}

// ---------------------------------------------------------------------------
// Flash attention, f16 WMMA (16x16x32), 32-key tiles, f32 softmax state.
// One wave = 16 query rows; 8 waves/block; no block barriers.
// ---------------------------------------------------------------------------
__global__ __launch_bounds__(256) void attn_f16_kernel(
    const _Float16* __restrict__ q16, const _Float16* __restrict__ k16,
    const _Float16* __restrict__ vt16, _Float16* __restrict__ y16) {
    __shared__ _Float16 pbuf[8][16][40];  // P tile; 80B row stride (16B aligned)

    const int lane = threadIdx.x & 31;
    const int wave = threadIdx.x >> 5;
    const int lane16 = lane & 15;
    const int half = lane >> 4;

    const int bh = blockIdx.x & 63;
    const int qblk = blockIdx.x >> 6;
    const int q0 = qblk * 128 + wave * 16;

    const _Float16* qp = q16 + (size_t)bh * T_ * D_;
    const _Float16* kp = k16 + (size_t)bh * T_ * D_;
    const _Float16* vtp = vt16 + (size_t)bh * T_ * D_;

    // Q A-fragments for the two 32-wide d-steps.
    F16Frag qf[2];
    const _Float16* qrow = qp + (size_t)(q0 + lane16) * D_;
#pragma unroll
    for (int s = 0; s < 2; ++s) {
        qf[s].q[0] = *(const uint4*)(qrow + 32 * s + half * 16);
        qf[s].q[1] = *(const uint4*)(qrow + 32 * s + half * 16 + 8);
    }

    v8f o[4];
#pragma unroll
    for (int j = 0; j < 4; ++j) o[j] = zero8();
    float rowmax[8], rowsum[8];
#pragma unroll
    for (int r = 0; r < 8; ++r) {
        rowmax[r] = -1e30f;
        rowsum[r] = 0.0f;
    }

    for (int key0 = 0; key0 <= q0 + 15; key0 += 32) {
        // S tiles for keys [key0,key0+16) and [key0+16,key0+32)
        v8f s0 = zero8(), s1 = zero8();
#pragma unroll
        for (int s = 0; s < 2; ++s) {
            const _Float16* kr0 =
                kp + (size_t)(key0 + lane16) * D_ + 32 * s + half * 16;
            F16Frag kb0, kb1;
            kb0.q[0] = *(const uint4*)(kr0);
            kb0.q[1] = *(const uint4*)(kr0 + 8);
            const _Float16* kr1 = kr0 + 16 * D_;
            kb1.q[0] = *(const uint4*)(kr1);
            kb1.q[1] = *(const uint4*)(kr1 + 8);
            s0 = wmma16(qf[s].v, kb0.v, s0);
            s1 = wmma16(qf[s].v, kb1.v, s1);
        }

        if (key0 + 31 > q0) {  // causal mask needed only near the diagonal
            int col0 = key0 + lane16;
            int col1 = col0 + 16;
#pragma unroll
            for (int r = 0; r < 8; ++r) {
                int row = q0 + r + 8 * half;
                if (col0 > row) s0[r] = -1e30f;
                if (col1 > row) s1[r] = -1e30f;
            }
        }

        float p0[8], p1[8], alpha[8];
#pragma unroll
        for (int r = 0; r < 8; ++r) {
            float m = fmaxf(s0[r], s1[r]);
            m = fmaxf(m, __shfl_xor(m, 1, 16));
            m = fmaxf(m, __shfl_xor(m, 2, 16));
            m = fmaxf(m, __shfl_xor(m, 4, 16));
            m = fmaxf(m, __shfl_xor(m, 8, 16));
            float mn = fmaxf(rowmax[r], m);
            alpha[r] = __expf(rowmax[r] - mn);
            rowmax[r] = mn;
            p0[r] = __expf(s0[r] - mn);
            p1[r] = __expf(s1[r] - mn);
            float ps = p0[r] + p1[r];
            ps += __shfl_xor(ps, 1, 16);
            ps += __shfl_xor(ps, 2, 16);
            ps += __shfl_xor(ps, 4, 16);
            ps += __shfl_xor(ps, 8, 16);
            rowsum[r] = rowsum[r] * alpha[r] + ps;
        }
#pragma unroll
        for (int j = 0; j < 4; ++j)
#pragma unroll
            for (int r = 0; r < 8; ++r) o[j][r] *= alpha[r];

        // P (16x32) f16 via per-wave LDS: C-layout -> A-layout.
#pragma unroll
        for (int r = 0; r < 8; ++r) {
            pbuf[wave][r + 8 * half][lane16] = (_Float16)p0[r];
            pbuf[wave][r + 8 * half][16 + lane16] = (_Float16)p1[r];
        }
        asm volatile("s_wait_dscnt 0" ::: "memory");

        F16Frag pa;
        pa.q[0] = *(const uint4*)&pbuf[wave][lane16][half * 16];
        pa.q[1] = *(const uint4*)&pbuf[wave][lane16][half * 16 + 8];

        // O += P * V : 4 WMMAs (one per 16-d slice), V^T fragments contiguous.
#pragma unroll
        for (int j = 0; j < 4; ++j) {
            const _Float16* vr =
                vtp + (size_t)(16 * j + lane16) * T_ + key0 + half * 16;
            F16Frag vb;
            vb.q[0] = *(const uint4*)(vr);
            vb.q[1] = *(const uint4*)(vr + 8);
            o[j] = wmma16(pa.v, vb.v, o[j]);
        }
    }

    const int b = bh >> 4;
    const int h = bh & 15;
#pragma unroll
    for (int j = 0; j < 4; ++j) {
#pragma unroll
        for (int r = 0; r < 8; ++r) {
            int row = q0 + r + 8 * half;
            int d = 16 * j + lane16;
            y16[(size_t)(b * T_ + row) * C_ + h * D_ + d] =
                (_Float16)(o[j][r] / rowsum[r]);
        }
    }
}

// ---------------------------------------------------------------------------
// Kernel: out(f32) = y16 @ W_proj
// ---------------------------------------------------------------------------
__global__ __launch_bounds__(256) void proj_gemm_f16(
    const _Float16* __restrict__ y16, const _Float16* __restrict__ WpT,
    float* __restrict__ out) {
    __shared__ _Float16 Bs[2 * 4096];

    const int lane = threadIdx.x & 31;
    const int wave = threadIdx.x >> 5;
    const int lane16 = lane & 15;
    const int half = lane >> 4;

    const int NB = C_ / 64;  // 16
    const int m0 = (blockIdx.x / NB) * 256 + wave * 32;
    const int n0 = (blockIdx.x % NB) * 64;

    v8f acc[2][4];
#pragma unroll
    for (int g = 0; g < 2; ++g)
#pragma unroll
        for (int j = 0; j < 4; ++j) acc[g][j] = zero8();

    gemm_f16_tile32(y16, WpT, m0, n0, Bs, acc);

#pragma unroll
    for (int g = 0; g < 2; ++g) {
#pragma unroll
        for (int j = 0; j < 4; ++j) {
#pragma unroll
            for (int r = 0; r < 8; ++r) {
                int row = m0 + g * 16 + r + 8 * half;
                int col = n0 + 16 * j + lane16;
                out[(size_t)row * C_ + col] = acc[g][j][r];
            }
        }
    }
}

extern "C" void kernel_launch(void* const* d_in, const int* in_sizes, int n_in,
                              void* d_out, int out_size, void* d_ws,
                              size_t ws_size, hipStream_t stream) {
    (void)in_sizes; (void)n_in; (void)out_size; (void)ws_size;
    const float* x = (const float*)d_in[0];   // [8192,1024]
    const float* Wa = (const float*)d_in[1];  // [1024,3072]
    const float* Wp = (const float*)d_in[2];  // [1024,1024]
    float* out = (float*)d_out;               // [8192,1024]

    const size_t NE = (size_t)B_ * H_ * T_ * D_;  // 8,388,608
    _Float16* x16 = (_Float16*)d_ws;
    _Float16* WaT = x16 + NE;                 // [3072][1024]
    _Float16* WpT = WaT + (size_t)N3C * C_;   // [1024][1024]
    _Float16* q16 = WpT + (size_t)C_ * C_;
    _Float16* k16 = q16 + NE;
    _Float16* vt16 = k16 + NE;
    _Float16* y16 = vt16 + NE;                // total ~92 MB

    cvt_f16_kernel<<<8192, 256, 0, stream>>>(x, x16);  // 8.4M elems
    cvt_transpose_kernel<<<16 * 48, 256, 0, stream>>>(Wa, WaT, C_, N3C);
    cvt_transpose_kernel<<<16 * 16, 256, 0, stream>>>(Wp, WpT, C_, C_);

    // qkv GEMM: 32 M-blocks * 48 N-blocks of 256x64
    qkv_gemm_f16<<<32 * 48, 256, 0, stream>>>(x16, WaT, q16, k16, vt16);
    // attention: 16 query blocks * 64 (b,h)
    attn_f16_kernel<<<1024, 256, 0, stream>>>(q16, k16, vt16, y16);
    // projection: 32 M-blocks * 16 N-blocks of 256x64
    proj_gemm_f16<<<32 * 16, 256, 0, stream>>>(y16, WpT, out);
}
